// CGRASVSR_74947179315519
// MI455X (gfx1250) — matrix operand order repaired and checked by reading, not compiled, not run on hardware
//
#include <hip/hip_runtime.h>
#include <math.h>

// ---------------------------------------------------------------------------
// CDNA5 (gfx1250) flow-guided deformable alignment.
// All GEMM-like work (4 convs + deformable-conv contraction) runs on
// v_wmma_f32_16x16x32_bf16. Activations are channels-last ([N][HW][C]) so a
// 16-element B fragment is two global_load_b128; weights are tap-major
// [O][9][CINP] (CINP % 32 == 0) so a whole 32-wide K step shares one 3x3 tap
// and the halo test is one branchless mask per step. fp32->bf16 uses the
// native hardware conversion (fptrunc), not software rounding.
// ---------------------------------------------------------------------------

typedef __attribute__((ext_vector_type(16))) __bf16    v16bf;
typedef __attribute__((ext_vector_type(8)))  float     v8f;
typedef __attribute__((ext_vector_type(4)))  unsigned  uint4e;
typedef __attribute__((ext_vector_type(4)))  float     float4e;

union BFrag { unsigned short us[16]; __bf16 b[16]; unsigned u32[8]; uint4e q[2]; v16bf v; };
union F8    { float f[8]; float4e q[2]; };
union H8    { unsigned short us[8]; __bf16 b[8]; uint4e q; };

__device__ __forceinline__ unsigned short f2bf_us(float f) {
  union { __bf16 b; unsigned short us; } u;
  u.b = (__bf16)f;                      // native v_cvt (RTNE)
  return u.us;
}

// ------------------------- pack / repack kernels ---------------------------

// feat = concat([extra_feat(192), flow_1(2), flow_2(2)]) zero-padded to 224
// channels, channels-last -> bf16 [N][HW][224]
__global__ void pack_feat(const float* __restrict__ ef,
                          const float* __restrict__ f1,
                          const float* __restrict__ f2,
                          unsigned short* __restrict__ outp,
                          int HW, int total) {
  int i = blockIdx.x * 256 + threadIdx.x;
  if (i >= total) return;
  int c = i % 224; int t = i / 224; int p = t % HW; int n = t / HW;
  float v = 0.f;
  if (c < 192)      v = ef[(n * 192 + c) * HW + p];
  else if (c < 194) v = f1[(n * 2 + (c - 192)) * HW + p];
  else if (c < 196) v = f2[(n * 2 + (c - 194)) * HW + p];
  outp[i] = f2bf_us(v);
}

// x NCHW fp32 [N][128][HW] -> channels-last fp32 [N][HW][128]
__global__ void x_to_cl(const float* __restrict__ in,
                        float* __restrict__ outp, int HW, int total) {
  int i = blockIdx.x * 256 + threadIdx.x;
  if (i >= total) return;
  int c = i % 128; int t = i / 128; int p = t % HW; int n = t / HW;
  outp[i] = in[((size_t)n * 128 + c) * HW + p];
}

// OIHW fp32 [O][CIN][9] -> tap-major bf16 [O][9][CINP], channels >= CIN zero.
__global__ void repack_w(const float* __restrict__ w,
                         unsigned short* __restrict__ outp,
                         int CIN, int CINP, int total) {
  int i = blockIdx.x * 256 + threadIdx.x;
  if (i >= total) return;
  int ci = i % CINP; int t = i / CINP; int tap = t % 9; int o = t / 9;
  float v = (ci < CIN) ? w[(o * CIN + ci) * 9 + tap] : 0.f;
  outp[i] = f2bf_us(v);
}

// conv_offset output (channels-last [N][HW][432]) -> {py,px,m,0} float4
// records [N][16][9][HW].
__global__ void make_offsets(const float* __restrict__ c3,
                             const float* __restrict__ f1,
                             const float* __restrict__ f2,
                             float4e* __restrict__ pym,
                             int H, int W, int total) {
  int i = blockIdx.x * 256 + threadIdx.x;
  if (i >= total) return;
  const int HW = H * W;
  int p = i % HW; int t = i / HW;
  int k = t % 9;  t /= 9;
  int g = t % 16; int n = t / 16;
  int yy = p / W, xx = p - yy * W;
  int cy = (g < 8) ? (g * 18 + 2 * k) : (144 + (g - 8) * 18 + 2 * k);
  const float* fl = (g < 8) ? f1 : f2;
  float fy = fl[(n * 2 + 1) * HW + p];   // flow.flip(1)
  float fx = fl[(n * 2 + 0) * HW + p];
  const float* c3p = c3 + ((size_t)n * HW + p) * 432;
  float oy = 10.f * tanhf(c3p[cy]) + fy + (float)(k / 3 - 1) + (float)yy;
  float ox = 10.f * tanhf(c3p[cy + 1]) + fx + (float)(k % 3 - 1) + (float)xx;
  float mm = 1.f / (1.f + __expf(-c3p[288 + g * 9 + k]));
  float4e r = {oy, ox, mm, 0.f};
  pym[((size_t)(n * 16 + g) * 9 + k) * HW + p] = r;
}

// ----------------- implicit-GEMM 3x3 SAME conv via WMMA bf16 ----------------
// Weights tap-major bf16 [O][9][CINP]; input channels-last bf16 [N][HW][CINP].
// Block = 128 threads = 4 waves; each wave owns 16 pixels and OCT 16-channel
// output tiles (B fragment reused across OCT WMMAs). B fragment = 2 b128.
template <int OCT>
__global__ void __launch_bounds__(128)
conv3x3_wmma(const unsigned short* __restrict__ Wt,
             const unsigned short* __restrict__ inBf,
             const float* __restrict__ bias,
             float* __restrict__ outF,          // channels-last fp32 (or null)
             unsigned short* __restrict__ outB, // channels-last bf16 (or null)
             int CINP, int O, int H, int W, int lrelu) {
  const int Ktot = 9 * CINP;
  const int HW = H * W;
  const int lane = threadIdx.x & 31;
  const int wave = threadIdx.x >> 5;
  const int col = lane & 15;
  const int hi = (lane >> 4) & 1;

  const int n = blockIdx.z;
  const int ocBase = blockIdx.y * (16 * OCT);
  const int p = (blockIdx.x << 6) + (wave << 4) + col;
  const int yy = p / W, xx = p - yy * W;

  const unsigned short* inN = inBf + (size_t)n * HW * CINP;
  const unsigned short* wBase = Wt + (size_t)(ocBase + col) * Ktot;

  v8f acc[OCT] = {};
  int tap = 0, crem = 0;
  for (int k0 = 0; k0 < Ktot; k0 += 32) {
    // one 3x3 tap covers the whole 32-wide K step (CINP % 32 == 0)
    int iy = yy + (tap / 3) - 1;
    int ix = xx + (tap - (tap / 3) * 3) - 1;
    bool valid = (iy >= 0) & (iy < H) & (ix >= 0) & (ix < W);
    int iyc = min(max(iy, 0), H - 1);
    int ixc = min(max(ix, 0), W - 1);
    int cb = crem + (hi ? 16 : 0);
    const unsigned short* src = inN + (size_t)(iyc * W + ixc) * CINP + cb;

    BFrag B;
    B.q[0] = *(const uint4e*)(src);        // 16 contiguous bf16 channels
    B.q[1] = *(const uint4e*)(src + 8);
    unsigned msk = valid ? 0xFFFFFFFFu : 0u;
#pragma unroll
    for (int i2 = 0; i2 < 8; ++i2) B.u32[i2] &= msk;   // branchless halo zero

#pragma unroll
    for (int t = 0; t < OCT; ++t) {
      const unsigned short* wR = wBase + (size_t)t * 16 * Ktot + k0;
      BFrag A;
#pragma unroll
      for (int j = 0; j < 8; ++j) {
        int kk = 2 * j + (hi ? 16 : 0);
        A.u32[j] = *(const unsigned*)(wR + kk);
      }
      acc[t] = __builtin_amdgcn_wmma_f32_16x16x32_bf16(false, A.v, false, B.v,
                                                       (short)0, acc[t], false, false);
    }
    if (k0 + 32 < Ktot) __builtin_prefetch(wBase + k0 + 32, 0, 1);
    crem += 32;
    if (crem == CINP) { crem = 0; ++tap; }
  }

  // D layout: VGPR j -> row M = j + (lane>=16?8:0), col = lane&15 (== pixel)
  if (outB) {
    unsigned short* dst = outB + ((size_t)n * HW + p) * O + ocBase + (hi ? 8 : 0);
#pragma unroll
    for (int t = 0; t < OCT; ++t) {
      H8 pk;
#pragma unroll
      for (int j = 0; j < 8; ++j) {
        int oc = ocBase + t * 16 + (hi ? 8 : 0) + j;
        float v = acc[t][j] + bias[oc];
        if (lrelu) v = (v >= 0.f) ? v : 0.1f * v;
        pk.b[j] = (__bf16)v;               // native cvt, pairs -> v_cvt_pk
      }
      *(uint4e*)(dst + t * 16) = pk.q;     // 8 channels, one b128 store
    }
  } else {
    float* dst = outF + ((size_t)n * HW + p) * O + ocBase + (hi ? 8 : 0);
#pragma unroll
    for (int t = 0; t < OCT; ++t) {
      F8 pk;
#pragma unroll
      for (int j = 0; j < 8; ++j) {
        int oc = ocBase + t * 16 + (hi ? 8 : 0) + j;
        float v = acc[t][j] + bias[oc];
        if (lrelu) v = (v >= 0.f) ? v : 0.1f * v;
        pk.f[j] = v;
      }
      *(float4e*)(dst + t * 16) = pk.q[0];
      *(float4e*)(dst + t * 16 + 4) = pk.q[1];
    }
  }
}

// -------- fused deformable conv: bilinear gather feeds WMMA B-matrix --------
// x is channels-last fp32 [N][HW][128]: each bilinear corner of a deformable
// group reads 8 contiguous channels (2 b128). Offsets+mask come as one float4
// record per (group,tap,pixel). Branchless clip-and-zero via bilinear weights.
__global__ void __launch_bounds__(128)
dcn_wmma(const unsigned short* __restrict__ Wt,   // [64][9][128] tap-major
         const float* __restrict__ xcl,           // [N][HW][128] fp32
         const float4e* __restrict__ pym,         // [N][16][9][HW] {py,px,m,0}
         const float* __restrict__ bias,
         float* __restrict__ out,                 // [N][64][H][W] (NCHW)
         int H, int W) {
  const int CINP = 128, O = 64, Ktot = 1152, OCT = 4;
  const int HW = H * W;
  const int lane = threadIdx.x & 31;
  const int wave = threadIdx.x >> 5;
  const int col = lane & 15;
  const int hi = (lane >> 4) & 1;

  const int n = blockIdx.z;
  const int p = (blockIdx.x << 6) + (wave << 4) + col;

  const float* xN = xcl + (size_t)n * HW * CINP;
  const float4e* pymN = pym + (size_t)n * 144 * HW;
  const unsigned short* wBase = Wt + (size_t)col * Ktot;

  v8f acc[OCT] = {};
  for (int k0 = 0; k0 < Ktot; k0 += 32) {
    int tap = k0 >> 7;                       // k0 / 128
    int cb = (k0 & 127) + (hi ? 16 : 0);     // first channel of this half

    BFrag B;
#pragma unroll
    for (int sub = 0; sub < 2; ++sub) {      // two deformable groups per half
      int g = (cb >> 3) + sub;
      float4e om = pymN[(size_t)(g * 9 + tap) * HW + p];
      float sy = om[0], sx = om[1], mv = om[2];
      float fy0 = floorf(sy), fx0 = floorf(sx);
      int iy = (int)fy0, ix = (int)fx0;
      float wy = sy - fy0, wx = sx - fx0;
      float vy0 = (iy >= 0 && iy < H) ? 1.f : 0.f;
      float vy1 = (iy + 1 >= 0 && iy + 1 < H) ? 1.f : 0.f;
      float vx0 = (ix >= 0 && ix < W) ? 1.f : 0.f;
      float vx1 = (ix + 1 >= 0 && ix + 1 < W) ? 1.f : 0.f;
      int iy0c = min(max(iy, 0), H - 1),     iy1c = min(max(iy + 1, 0), H - 1);
      int ix0c = min(max(ix, 0), W - 1),     ix1c = min(max(ix + 1, 0), W - 1);
      float w00 = (1.f - wy) * (1.f - wx) * vy0 * vx0 * mv;
      float w01 = (1.f - wy) * wx * vy0 * vx1 * mv;
      float w10 = wy * (1.f - wx) * vy1 * vx0 * mv;
      float w11 = wy * wx * vy1 * vx1 * mv;
      const float* xb = xN + cb + sub * 8;
      const float* c00 = xb + (size_t)(iy0c * W + ix0c) * CINP;
      const float* c01 = xb + (size_t)(iy0c * W + ix1c) * CINP;
      const float* c10 = xb + (size_t)(iy1c * W + ix0c) * CINP;
      const float* c11 = xb + (size_t)(iy1c * W + ix1c) * CINP;
      F8 v00, v01, v10, v11;
      v00.q[0] = *(const float4e*)(c00); v00.q[1] = *(const float4e*)(c00 + 4);
      v01.q[0] = *(const float4e*)(c01); v01.q[1] = *(const float4e*)(c01 + 4);
      v10.q[0] = *(const float4e*)(c10); v10.q[1] = *(const float4e*)(c10 + 4);
      v11.q[0] = *(const float4e*)(c11); v11.q[1] = *(const float4e*)(c11 + 4);
#pragma unroll
      for (int c = 0; c < 8; ++c) {
        float v = v00.f[c] * w00 + v01.f[c] * w01 + v10.f[c] * w10 + v11.f[c] * w11;
        B.b[sub * 8 + c] = (__bf16)v;        // native cvt, pairs -> v_cvt_pk
      }
    }
#pragma unroll
    for (int t = 0; t < OCT; ++t) {
      const unsigned short* wR = wBase + (size_t)t * 16 * Ktot + k0;
      BFrag A;
#pragma unroll
      for (int j = 0; j < 8; ++j) {
        int kk = 2 * j + (hi ? 16 : 0);
        A.u32[j] = *(const unsigned*)(wR + kk);
      }
      acc[t] = __builtin_amdgcn_wmma_f32_16x16x32_bf16(false, A.v, false, B.v,
                                                       (short)0, acc[t], false, false);
    }
  }
#pragma unroll
  for (int t = 0; t < OCT; ++t) {
#pragma unroll
    for (int j = 0; j < 8; ++j) {
      int oc = t * 16 + j + (hi ? 8 : 0);
      out[((size_t)n * O + oc) * HW + p] = acc[t][j] + bias[oc];
    }
  }
}

// ---------------------------------------------------------------------------

extern "C" void kernel_launch(void* const* d_in, const int* in_sizes, int n_in,
                              void* d_out, int out_size, void* d_ws, size_t ws_size,
                              hipStream_t stream) {
  (void)in_sizes; (void)n_in; (void)out_size; (void)ws_size;
  const int N = 4, H = 96, W = 96;
  const int HW = H * W;

  const float* x  = (const float*)d_in[0];
  const float* ef = (const float*)d_in[1];
  const float* f1 = (const float*)d_in[2];
  const float* f2 = (const float*)d_in[3];
  const float* w0 = (const float*)d_in[4];  const float* b0 = (const float*)d_in[5];
  const float* w1 = (const float*)d_in[6];  const float* b1 = (const float*)d_in[7];
  const float* w2 = (const float*)d_in[8];  const float* b2 = (const float*)d_in[9];
  const float* w3 = (const float*)d_in[10]; const float* b3 = (const float*)d_in[11];
  const float* wd = (const float*)d_in[12]; const float* bd = (const float*)d_in[13];
  float* out = (float*)d_out;

  // bump-allocate scratch from d_ws (~195 MB total)
  char* wsb = (char*)d_ws;
  size_t off = 0;
  auto alloc = [&](size_t bytes) -> char* {
    char* pp = wsb + off;
    off = (off + bytes + 255) & ~(size_t)255;
    return pp;
  };
  unsigned short* featBf = (unsigned short*)alloc((size_t)N * HW * 224 * 2);
  unsigned short* w0Bf = (unsigned short*)alloc((size_t)64 * 9 * 224 * 2);
  unsigned short* w1Bf = (unsigned short*)alloc((size_t)64 * 9 * 64 * 2);
  unsigned short* w2Bf = (unsigned short*)alloc((size_t)64 * 9 * 64 * 2);
  unsigned short* w3Bf = (unsigned short*)alloc((size_t)432 * 9 * 64 * 2);
  unsigned short* wdBf = (unsigned short*)alloc((size_t)64 * 9 * 128 * 2);
  unsigned short* hBfA = (unsigned short*)alloc((size_t)N * HW * 64 * 2);
  unsigned short* hBfB = (unsigned short*)alloc((size_t)N * HW * 64 * 2);
  float*   c3   = (float*)alloc((size_t)N * HW * 432 * 4);
  float*   xcl  = (float*)alloc((size_t)N * HW * 128 * 4);
  float4e* pymB = (float4e*)alloc((size_t)N * 144 * HW * 16);

  auto nblk = [](int t) { return dim3((unsigned)((t + 255) / 256)); };

  // packs / repacks
  int featTot = N * HW * 224;
  pack_feat<<<nblk(featTot), 256, 0, stream>>>(ef, f1, f2, featBf, HW, featTot);
  x_to_cl<<<nblk(N * HW * 128), 256, 0, stream>>>(x, xcl, HW, N * HW * 128);
  repack_w<<<nblk(64 * 9 * 224), 256, 0, stream>>>(w0, w0Bf, 196, 224, 64 * 9 * 224);
  repack_w<<<nblk(64 * 9 * 64), 256, 0, stream>>>(w1, w1Bf, 64, 64, 64 * 9 * 64);
  repack_w<<<nblk(64 * 9 * 64), 256, 0, stream>>>(w2, w2Bf, 64, 64, 64 * 9 * 64);
  repack_w<<<nblk(432 * 9 * 64), 256, 0, stream>>>(w3, w3Bf, 64, 64, 432 * 9 * 64);
  repack_w<<<nblk(64 * 9 * 128), 256, 0, stream>>>(wd, wdBf, 128, 128, 64 * 9 * 128);

  const dim3 cgrid64(HW / 64, 1, N);            // O=64, OCT=4
  const dim3 cgrid432(HW / 64, 432 / 48, N);    // O=432, OCT=3 -> grid.y=9

  // offset head (bf16 channels-last activations written from conv epilogue)
  conv3x3_wmma<4><<<cgrid64, 128, 0, stream>>>(w0Bf, featBf, b0, nullptr, hBfA,
                                               224, 64, H, W, 1);
  conv3x3_wmma<4><<<cgrid64, 128, 0, stream>>>(w1Bf, hBfA, b1, nullptr, hBfB,
                                               64, 64, H, W, 1);
  conv3x3_wmma<4><<<cgrid64, 128, 0, stream>>>(w2Bf, hBfB, b2, nullptr, hBfA,
                                               64, 64, H, W, 1);
  conv3x3_wmma<3><<<cgrid432, 128, 0, stream>>>(w3Bf, hBfA, b3, c3, nullptr,
                                                64, 432, H, W, 0);

  // offsets + mask -> float4 records
  int offTot = N * 16 * 9 * HW;
  make_offsets<<<nblk(offTot), 256, 0, stream>>>(c3, f1, f2, pymB, H, W, offTot);

  // fused deformable conv
  dcn_wmma<<<cgrid64, 128, 0, stream>>>(wdBf, xcl, pymB, bd, out, H, W);
}
